// Qwen3NextSparseMoeBlock_24687472017510
// MI455X (gfx1250) — matrix operand order repaired and checked by reading, not compiled, not run on hardware
//
#include <hip/hip_runtime.h>
#include <stdint.h>

// ---------------------------------------------------------------------------
// Qwen3-Next sparse MoE block, MI455X (gfx1250, wave32, WMMA bf16).
//   T=1024 tokens, H=2048, E=64 experts, top-8, I=SI=512.
// Roofline: expert weights (805 MB fp32) stream once from HBM (~35us @ 23.3TB/s);
// sparse compute (~58 GFLOP) runs on v_wmma_f32_16x16x32_bf16 well under that.
// Per block: one expert, 32 output columns; 8 waves cover 8 M-tiles (128
// gathered token rows). Weight K-slices (32x32 fp32) are loaded non-temporal,
// converted to bf16, staged into LDS in WMMA-B fragment order, double-buffered.
// The inner k-loop is branch-free (last step peeled, compute unconditional,
// stores guarded) so the pre-WMMA wait is precise and the weight stream stays
// in flight under the WMMAs.
// ---------------------------------------------------------------------------

typedef __attribute__((ext_vector_type(16))) __bf16 v16bf;
typedef __attribute__((ext_vector_type(2)))  __bf16 v2bf;
typedef __attribute__((ext_vector_type(8)))  float  v8f;

constexpr int HD = 2048;          // hidden
constexpr int NE = 64;            // experts
constexpr int IM = 512;           // intermediate (== shared intermediate)
constexpr int NT = 1024;          // tokens (B*S)
constexpr int SH_ROW_BASE = 9216; // hdn row base for shared expert (>= 8192+64*15)
constexpr int HDN_ROWS = SH_ROW_BASE + NT;

// workspace layout (bytes)
constexpr size_t OFF_XB   = 0;                                   // NT*HD bf16
constexpr size_t OFF_HDN  = OFF_XB  + (size_t)NT * HD * 2;       // HDN_ROWS*IM bf16
constexpr size_t OFF_CNT  = OFF_HDN + (size_t)HDN_ROWS * IM * 2; // NE ints
constexpr size_t OFF_BASE = OFF_CNT + 256;                       // NE ints
constexpr size_t OFF_LTOK = OFF_BASE + 256;                      // NE*NT ints
constexpr size_t OFF_LWT  = OFF_LTOK + (size_t)NE * NT * 4;      // NE*NT floats
constexpr size_t OFF_SG   = OFF_LWT  + (size_t)NE * NT * 4;      // NT floats
constexpr size_t WS_NEED  = OFF_SG + (size_t)NT * 4;

__device__ __forceinline__ unsigned short f2bf(float f) {
  unsigned int u = __float_as_uint(f);
  unsigned int r = u + 0x7FFFu + ((u >> 16) & 1u);   // round-to-nearest-even
  return (unsigned short)(r >> 16);
}
__device__ __forceinline__ unsigned int packbf(float lo, float hi) {
#if __has_builtin(__builtin_amdgcn_cvt_pk_bf16_f32)
  union { v2bf v; unsigned int u; } cv;
  cv.v = __builtin_amdgcn_cvt_pk_bf16_f32(lo, hi);
  return cv.u;
#else
  return (unsigned int)f2bf(lo) | ((unsigned int)f2bf(hi) << 16);
#endif
}

union FragBF { v16bf v; uint4 q[2]; };

__device__ __forceinline__ v8f wmma_bf16(v16bf a, v16bf b, v8f c) {
  return __builtin_amdgcn_wmma_f32_16x16x32_bf16(false, a, false, b,
                                                 (short)0, c, false, false);
}
__device__ __forceinline__ v8f vzero8() {
  v8f z = {0.f, 0.f, 0.f, 0.f, 0.f, 0.f, 0.f, 0.f};
  return z;
}

// LDS u16 index of element (k, c) of a 32(K) x 32(N) slice stored as two
// 512-entry B-fragments (sub-tile = c>>4): lane = (c&15) + ((k&8)<<1), elem = (k&7) | ((k&16)>>1).
__device__ __forceinline__ int bidx(int k, int c) {
  return ((c >> 4) << 9) + (((c & 15) + ((k & 8) << 1)) << 4) + ((k & 7) | ((k & 16) >> 1));
}

// ---------------------------------------------------------------------------
// Router: logits, top-8 softmax (== renormalized top-k of the full softmax),
// sigmoid shared gate, x -> bf16, scatter tokens into per-expert lists.
// One block per token.
// ---------------------------------------------------------------------------
__global__ __launch_bounds__(256) void router_kernel(
    const float* __restrict__ x, const float* __restrict__ gw,
    const float* __restrict__ segw,
    unsigned short* __restrict__ xb,
    int* __restrict__ cnt, int* __restrict__ ltok, float* __restrict__ lwt,
    float* __restrict__ sg)
{
  const int t = blockIdx.x;
  const int tid = threadIdx.x;
  __shared__ float sred[256];
  __shared__ float slog[NE];

  const float* xt = x + (size_t)t * HD;

  {
    const int e = tid & 63, chunk = tid >> 6;
    float p = 0.f;
    const int h0 = chunk * 512;
    for (int h = h0; h < h0 + 512; ++h) p += xt[h] * gw[h * NE + e];
    sred[tid] = p;
  }
  __syncthreads();
  if (tid < NE)
    slog[tid] = sred[tid] + sred[tid + 64] + sred[tid + 128] + sred[tid + 192];

  for (int h = tid; h < HD; h += 256) xb[(size_t)t * HD + h] = f2bf(xt[h]);

  float q = 0.f;
  for (int h = tid; h < HD; h += 256) q += xt[h] * segw[h];
  __syncthreads();
  sred[tid] = q;
  __syncthreads();
  for (int s = 128; s > 0; s >>= 1) {
    if (tid < s) sred[tid] += sred[tid + s];
    __syncthreads();
  }

  if (tid == 0) {
    sg[t] = 1.f / (1.f + __expf(-sred[0]));
    unsigned long long used = 0ull;
    int   sel[8];
    float lsel[8];
    for (int j = 0; j < 8; ++j) {
      int best = 0; float bv = -3.4e38f;
      for (int e2 = 0; e2 < NE; ++e2) {
        if (used & (1ull << e2)) continue;
        float v = slog[e2];
        if (v > bv) { bv = v; best = e2; }
      }
      used |= (1ull << best);
      sel[j] = best; lsel[j] = bv;
    }
    const float mx = lsel[0];
    float wv[8], s = 0.f;
    for (int j = 0; j < 8; ++j) { wv[j] = __expf(lsel[j] - mx); s += wv[j]; }
    const float inv = 1.f / s;
    for (int j = 0; j < 8; ++j) {
      const int e2 = sel[j];
      const int slot = atomicAdd(&cnt[e2], 1);
      ltok[e2 * NT + slot] = t;
      lwt[e2 * NT + slot]  = wv[j] * inv;
    }
  }
}

// ---------------------------------------------------------------------------
// Prefix over padded (to 16) per-expert counts + zero-fill pad slots.
// ---------------------------------------------------------------------------
__global__ void prefix_kernel(const int* __restrict__ cnt, int* __restrict__ baseRows,
                              int* __restrict__ ltok, float* __restrict__ lwt)
{
  if (threadIdx.x == 0) {
    int acc = 0;
    for (int e = 0; e < NE; ++e) {
      baseRows[e] = acc;
      acc += (cnt[e] + 15) & ~15;
    }
  }
  __syncthreads();
  for (int e = 0; e < NE; ++e) {
    const int c = cnt[e];
    const int rp = (c + 15) & ~15;
    for (int p = c + threadIdx.x; p < rp; p += blockDim.x) {
      ltok[e * NT + p] = 0;     // valid address, weight 0 -> contributes nothing
      lwt[e * NT + p] = 0.f;
    }
  }
}

// ---------------------------------------------------------------------------
// Gate+Up SwiGLU GEMM. Block: (expert, 32-column slice). K = 2048 step 32.
// Double-buffered LDS staging; each wave does 4 WMMAs per k-step
// (gate/up x 2 N-subtiles) against one shared A-fragment load.
// blockIdx.y == 64 -> shared expert (identity token map, all 1024 rows).
// ---------------------------------------------------------------------------
__global__ __launch_bounds__(256) void moe_gateup_kernel(
    const unsigned short* __restrict__ xb,
    const float* __restrict__ eg, const float* __restrict__ eu,
    const float* __restrict__ sgw, const float* __restrict__ suw,
    const int* __restrict__ cnt, const int* __restrict__ baseRows,
    const int* __restrict__ ltok,
    unsigned short* __restrict__ hdn)
{
  const int nblk = blockIdx.x;          // 0..15 (32 cols each)
  const int e    = blockIdx.y;          // 0..64 (64 == shared expert)
  const int tid  = threadIdx.x;
  const int lane = tid & 31;
  const int wv   = tid >> 5;
  const int n0   = nblk * 32;

  const float* Wg; const float* Wu; const int* lt; int rowsPad, hbase;
  if (e < NE) {
    Wg = eg + (size_t)e * HD * IM;
    Wu = eu + (size_t)e * HD * IM;
    rowsPad = (cnt[e] + 15) & ~15;
    hbase = baseRows[e];
    lt = ltok + e * NT;
  } else {
    Wg = sgw; Wu = suw; rowsPad = NT; hbase = SH_ROW_BASE; lt = nullptr;
  }

  __shared__ __align__(16) unsigned short sG[2][1024];
  __shared__ __align__(16) unsigned short sU[2][1024];

  // staging: thread stages K-pairs (2*kp0, 2*kp0+1) and (2*kp0+16, +17) at col c0
  const int c0  = tid & 31;
  const int kp0 = tid >> 5;             // 0..7
  const int i0  = bidx(2 * kp0,      c0);  // even elem -> u32 store
  const int i1  = bidx(2 * kp0 + 16, c0);
  const float* WgT = Wg + n0 + c0;
  const float* WuT = Wu + n0 + c0;
  const size_t r0 = (size_t)(2 * kp0) * IM;
  const size_t r1 = (size_t)(2 * kp0 + 16) * IM;

  const int mgroups = (rowsPad + 127) >> 7;
  for (int mg = 0; mg < mgroups; ++mg) {
    const int mtile = mg * 8 + wv;
    const bool active = (mtile * 16) < rowsPad;   // wave-uniform
    // Inactive waves clamp to row 0 and compute (discarded) results so the
    // hot loop has no branches; only the epilogue store is guarded.
    const int p   = active ? (mtile * 16 + (lane & 15)) : 0;
    const int tok = lt ? lt[p] : p;
    const size_t aoff = (size_t)tok * HD + ((lane & 16) ? 8 : 0);

    v8f aG0 = vzero8(), aG1 = vzero8(), aU0 = vzero8(), aU1 = vzero8();

    // prologue: stage k-step 0 into buffer 0 (non-temporal weight stream)
    {
      const float g0 = __builtin_nontemporal_load(&WgT[r0]);
      const float g1 = __builtin_nontemporal_load(&WgT[r0 + IM]);
      const float g2 = __builtin_nontemporal_load(&WgT[r1]);
      const float g3 = __builtin_nontemporal_load(&WgT[r1 + IM]);
      const float u0 = __builtin_nontemporal_load(&WuT[r0]);
      const float u1 = __builtin_nontemporal_load(&WuT[r0 + IM]);
      const float u2 = __builtin_nontemporal_load(&WuT[r1]);
      const float u3 = __builtin_nontemporal_load(&WuT[r1 + IM]);
      *(unsigned int*)&sG[0][i0] = packbf(g0, g1);
      *(unsigned int*)&sG[0][i1] = packbf(g2, g3);
      *(unsigned int*)&sU[0][i0] = packbf(u0, u1);
      *(unsigned int*)&sU[0][i1] = packbf(u2, u3);
    }
    __syncthreads();

    int buf = 0;
    // main loop: branch-free body; last k-step peeled below
    for (int kk = 0; kk < HD - 32; kk += 32) {
      FragBF A;
      A.q[0] = *(const uint4*)(xb + aoff + kk);        // K = kk+{0..7|8..15}
      A.q[1] = *(const uint4*)(xb + aoff + kk + 16);   // K = kk+{16..23|24..31}

      const size_t ro = (size_t)(kk + 32) * IM;
      const float g0 = __builtin_nontemporal_load(&WgT[ro + r0]);
      const float g1 = __builtin_nontemporal_load(&WgT[ro + r0 + IM]);
      const float g2 = __builtin_nontemporal_load(&WgT[ro + r1]);
      const float g3 = __builtin_nontemporal_load(&WgT[ro + r1 + IM]);
      const float u0 = __builtin_nontemporal_load(&WuT[ro + r0]);
      const float u1 = __builtin_nontemporal_load(&WuT[ro + r0 + IM]);
      const float u2 = __builtin_nontemporal_load(&WuT[ro + r1]);
      const float u3 = __builtin_nontemporal_load(&WuT[ro + r1 + IM]);

      FragBF B;
      const uint4* bp;
      bp = (const uint4*)&sG[buf][lane << 4];         B.q[0] = bp[0]; B.q[1] = bp[1];
      aG0 = wmma_bf16(A.v, B.v, aG0);
      bp = (const uint4*)&sG[buf][512 + (lane << 4)]; B.q[0] = bp[0]; B.q[1] = bp[1];
      aG1 = wmma_bf16(A.v, B.v, aG1);
      bp = (const uint4*)&sU[buf][lane << 4];         B.q[0] = bp[0]; B.q[1] = bp[1];
      aU0 = wmma_bf16(A.v, B.v, aU0);
      bp = (const uint4*)&sU[buf][512 + (lane << 4)]; B.q[0] = bp[0]; B.q[1] = bp[1];
      aU1 = wmma_bf16(A.v, B.v, aU1);

      const int nb = buf ^ 1;
      *(unsigned int*)&sG[nb][i0] = packbf(g0, g1);
      *(unsigned int*)&sG[nb][i1] = packbf(g2, g3);
      *(unsigned int*)&sU[nb][i0] = packbf(u0, u1);
      *(unsigned int*)&sU[nb][i1] = packbf(u2, u3);
      __syncthreads();
      buf ^= 1;
    }
    // peeled last k-step (no staging)
    {
      const int kk = HD - 32;
      FragBF A;
      A.q[0] = *(const uint4*)(xb + aoff + kk);
      A.q[1] = *(const uint4*)(xb + aoff + kk + 16);
      FragBF B;
      const uint4* bp;
      bp = (const uint4*)&sG[buf][lane << 4];         B.q[0] = bp[0]; B.q[1] = bp[1];
      aG0 = wmma_bf16(A.v, B.v, aG0);
      bp = (const uint4*)&sG[buf][512 + (lane << 4)]; B.q[0] = bp[0]; B.q[1] = bp[1];
      aG1 = wmma_bf16(A.v, B.v, aG1);
      bp = (const uint4*)&sU[buf][lane << 4];         B.q[0] = bp[0]; B.q[1] = bp[1];
      aU0 = wmma_bf16(A.v, B.v, aU0);
      bp = (const uint4*)&sU[buf][512 + (lane << 4)]; B.q[0] = bp[0]; B.q[1] = bp[1];
      aU1 = wmma_bf16(A.v, B.v, aU1);
    }

    if (active) {
      // C/D layout: VGPR r -> M = r (+8 for lanes 16..31), N = lane&15
      const int rb = hbase + mtile * 16 + ((lane & 16) ? 8 : 0);
      const int cl = lane & 15;
      for (int r = 0; r < 8; ++r) {
        float g = aG0[r], u = aU0[r];
        hdn[(size_t)(rb + r) * IM + n0 + cl] = f2bf((g / (1.f + __expf(-g))) * u);
        g = aG1[r]; u = aU1[r];
        hdn[(size_t)(rb + r) * IM + n0 + 16 + cl] = f2bf((g / (1.f + __expf(-g))) * u);
      }
    }
    // next m-group's prologue writes buffer 0, whose last readers completed
    // before the final in-loop barrier; its own barrier orders the rest.
  }
}

// ---------------------------------------------------------------------------
// Down projection: hdn[rows, 512] @ dw[512, 2048], scaled by routing weight
// (sigmoid gate for the shared expert), atomically accumulated into out.
// Block: (expert, 32-column slice of H). K = 512 step 32.
// ---------------------------------------------------------------------------
__global__ __launch_bounds__(256) void moe_down_kernel(
    const unsigned short* __restrict__ hdn,
    const float* __restrict__ ed, const float* __restrict__ sdw,
    const int* __restrict__ cnt, const int* __restrict__ baseRows,
    const int* __restrict__ ltok, const float* __restrict__ lwt,
    const float* __restrict__ sg, float* __restrict__ out)
{
  const int nblk = blockIdx.x;          // 0..63 (32 cols each)
  const int e    = blockIdx.y;          // 0..64
  const int tid  = threadIdx.x;
  const int lane = tid & 31;
  const int wv   = tid >> 5;
  const int n0   = nblk * 32;

  const float* Wd; int rowsPad, hbase;
  if (e < NE) {
    Wd = ed + (size_t)e * IM * HD;
    rowsPad = (cnt[e] + 15) & ~15;
    hbase = baseRows[e];
  } else {
    Wd = sdw; rowsPad = NT; hbase = SH_ROW_BASE;
  }

  __shared__ __align__(16) unsigned short sB[2][1024];

  const int c0  = tid & 31;
  const int kp0 = tid >> 5;
  const int i0  = bidx(2 * kp0,      c0);
  const int i1  = bidx(2 * kp0 + 16, c0);
  const float* WdT = Wd + n0 + c0;
  const size_t r0 = (size_t)(2 * kp0) * HD;
  const size_t r1 = (size_t)(2 * kp0 + 16) * HD;

  const int mgroups = (rowsPad + 127) >> 7;
  for (int mg = 0; mg < mgroups; ++mg) {
    const int mtile = mg * 8 + wv;
    const bool active = (mtile * 16) < rowsPad;
    const int p = active ? (mtile * 16 + (lane & 15)) : 0;
    const size_t aoff = (size_t)(hbase + p) * IM + ((lane & 16) ? 8 : 0);

    v8f a0 = vzero8(), a1 = vzero8();

    {
      const float w0 = __builtin_nontemporal_load(&WdT[r0]);
      const float w1 = __builtin_nontemporal_load(&WdT[r0 + HD]);
      const float w2 = __builtin_nontemporal_load(&WdT[r1]);
      const float w3 = __builtin_nontemporal_load(&WdT[r1 + HD]);
      *(unsigned int*)&sB[0][i0] = packbf(w0, w1);
      *(unsigned int*)&sB[0][i1] = packbf(w2, w3);
    }
    __syncthreads();

    int buf = 0;
    for (int kk = 0; kk < IM - 32; kk += 32) {
      FragBF A;
      A.q[0] = *(const uint4*)(hdn + aoff + kk);
      A.q[1] = *(const uint4*)(hdn + aoff + kk + 16);

      const size_t ro = (size_t)(kk + 32) * HD;
      const float w0 = __builtin_nontemporal_load(&WdT[ro + r0]);
      const float w1 = __builtin_nontemporal_load(&WdT[ro + r0 + HD]);
      const float w2 = __builtin_nontemporal_load(&WdT[ro + r1]);
      const float w3 = __builtin_nontemporal_load(&WdT[ro + r1 + HD]);

      FragBF B;
      const uint4* bp;
      bp = (const uint4*)&sB[buf][lane << 4];         B.q[0] = bp[0]; B.q[1] = bp[1];
      a0 = wmma_bf16(A.v, B.v, a0);
      bp = (const uint4*)&sB[buf][512 + (lane << 4)]; B.q[0] = bp[0]; B.q[1] = bp[1];
      a1 = wmma_bf16(A.v, B.v, a1);

      const int nb = buf ^ 1;
      *(unsigned int*)&sB[nb][i0] = packbf(w0, w1);
      *(unsigned int*)&sB[nb][i1] = packbf(w2, w3);
      __syncthreads();
      buf ^= 1;
    }
    {
      const int kk = IM - 32;
      FragBF A;
      A.q[0] = *(const uint4*)(hdn + aoff + kk);
      A.q[1] = *(const uint4*)(hdn + aoff + kk + 16);
      FragBF B;
      const uint4* bp;
      bp = (const uint4*)&sB[buf][lane << 4];         B.q[0] = bp[0]; B.q[1] = bp[1];
      a0 = wmma_bf16(A.v, B.v, a0);
      bp = (const uint4*)&sB[buf][512 + (lane << 4)]; B.q[0] = bp[0]; B.q[1] = bp[1];
      a1 = wmma_bf16(A.v, B.v, a1);
    }

    if (active) {
      const int pbase = mtile * 16 + ((lane & 16) ? 8 : 0);
      const int cl = lane & 15;
      for (int r = 0; r < 8; ++r) {
        const int pr = pbase + r;
        int tok; float w;
        if (e < NE) { tok = ltok[e * NT + pr]; w = lwt[e * NT + pr]; }
        else        { tok = pr;                w = sg[pr]; }
        atomicAdd(&out[(size_t)tok * HD + n0 + cl],      w * a0[r]);
        atomicAdd(&out[(size_t)tok * HD + n0 + 16 + cl], w * a1[r]);
      }
    }
  }
}

// ---------------------------------------------------------------------------
extern "C" void kernel_launch(void* const* d_in, const int* in_sizes, int n_in,
                              void* d_out, int out_size, void* d_ws, size_t ws_size,
                              hipStream_t stream)
{
  (void)in_sizes; (void)n_in; (void)out_size;
  const float* x    = (const float*)d_in[0];  // [1,1024,2048]
  const float* gw   = (const float*)d_in[1];  // [2048,64]
  const float* egw  = (const float*)d_in[2];  // [64,2048,512]
  const float* euw  = (const float*)d_in[3];  // [64,2048,512]
  const float* edw  = (const float*)d_in[4];  // [64,512,2048]
  const float* sgw  = (const float*)d_in[5];  // [2048,512]
  const float* suw  = (const float*)d_in[6];  // [2048,512]
  const float* sdw  = (const float*)d_in[7];  // [512,2048]
  const float* segw = (const float*)d_in[8];  // [2048,1]
  float* out = (float*)d_out;
  char*  ws  = (char*)d_ws;
  if (ws_size < WS_NEED) return;

  unsigned short* xb   = (unsigned short*)(ws + OFF_XB);
  unsigned short* hdn  = (unsigned short*)(ws + OFF_HDN);
  int*            cnt  = (int*)(ws + OFF_CNT);
  int*            base = (int*)(ws + OFF_BASE);
  int*            ltok = (int*)(ws + OFF_LTOK);
  float*          lwt  = (float*)(ws + OFF_LWT);
  float*          sg   = (float*)(ws + OFF_SG);

  hipMemsetAsync(cnt, 0, NE * sizeof(int), stream);
  hipMemsetAsync(out, 0, (size_t)NT * HD * sizeof(float), stream);

  router_kernel<<<NT, 256, 0, stream>>>(x, gw, segw, xb, cnt, ltok, lwt, sg);
  prefix_kernel<<<1, 256, 0, stream>>>(cnt, base, ltok, lwt);
  moe_gateup_kernel<<<dim3(IM / 32, NE + 1), 256, 0, stream>>>(
      xb, egw, euw, sgw, suw, cnt, base, ltok, hdn);
  moe_down_kernel<<<dim3(HD / 32, NE + 1), 256, 0, stream>>>(
      hdn, edw, sdw, cnt, base, ltok, lwt, sg, out);
}
